// KANLayer_5411658793270
// MI455X (gfx1250) — matrix operand order repaired and checked by reading, not compiled, not run on hardware
//
#include <hip/hip_runtime.h>
#include <hip/hip_bf16.h>
#include <math.h>

typedef __attribute__((ext_vector_type(16))) _Float16     v16h;
typedef __attribute__((ext_vector_type(8)))  float        v8f;
typedef __attribute__((ext_vector_type(2)))  __fp16       fp16x2;
typedef __attribute__((ext_vector_type(8)))  unsigned int v8u;
typedef __attribute__((ext_vector_type(4)))  unsigned int u32x4;
typedef __attribute__((ext_vector_type(8)))  int          i32x8;
typedef __attribute__((ext_vector_type(4)))  int          i32x4;

#define BATCH        32768
#define IN_F         512
#define OUT_F        64
#define NB           8
#define KT_PER_STAGE 18          // 16 spline k-tiles + 2 base (silu) k-tiles
#define N_STAGES     8           // 8 stages x 64 features = 512
#define N_KT         (KT_PER_STAGE * N_STAGES)     // 144 k-tiles of K=32
#define ROWS_PER_BLK 128
#define XS           68          // LDS row stride (floats): 64 + 4 TDM pad
#define CHUNK_HALVES (KT_PER_STAGE * 4 * 32 * 16)  // 36864 f16 per stage

// ---------------------------------------------------------------------------
// Prep kernel: pack (scale*spline_weight | base_weight) into per-lane WMMA
// B-fragment layout (f16). B 32x16: lanes 0-15 K=e<8?e:e+8, lanes 16-31 K=e<8?8+e:16+e.
// ---------------------------------------------------------------------------
__global__ void kan_pack_w(const float* __restrict__ sw,
                           const float* __restrict__ bw,
                           const float* __restrict__ ss,
                           _Float16* __restrict__ wpack) {
  int idx = blockIdx.x * blockDim.x + threadIdx.x;
  const int total = N_KT * 4 * 32 * 16;
  if (idx >= total) return;
  int e    = idx & 15;
  int lane = (idx >> 4) & 31;
  int nt   = (idx >> 9) & 3;
  int kt   = idx >> 11;
  int n    = nt * 16 + (lane & 15);
  int Koff = ((lane >> 4) << 3) + (e & 7) + ((e >> 3) << 4);
  int s = kt / KT_PER_STAGE;
  int t = kt % KT_PER_STAGE;
  float val;
  if (t < 16) {
    int i  = s * 64 + t * 4 + (Koff >> 3);
    int nb = Koff & 7;
    val = ss[i * OUT_F + n] * sw[(i * OUT_F + n) * NB + nb];
  } else {
    int i = s * 64 + (t - 16) * 32 + Koff;
    val = bw[i * OUT_F + n];
  }
  wpack[idx] = (_Float16)val;
}

// ---------------------------------------------------------------------------
// f16-pair pack (v_cvt_pkrtz_f16_f32)
// ---------------------------------------------------------------------------
__device__ __forceinline__ unsigned pkh(float a, float b) {
  union { fp16x2 h; unsigned u; } cv;
  cv.h = __builtin_amdgcn_cvt_pkrtz(a, b);
  return cv.u;
}

__device__ __forceinline__ float silu(float v) {
  return v / (1.0f + __expf(-v));
}

// ---------------------------------------------------------------------------
// Local de Boor cubic basis, uniform knots t[j] = -2.2 + 0.4j.  Produces the
// 8-slot basis row directly as 4 packed f16 pairs (pair-granularity scatter).
// ---------------------------------------------------------------------------
__device__ __forceinline__ void basis8_pk(float x, unsigned p[4]) {
  float xc = fminf(fmaxf(x, -1.0f), 1.0f);
  int c = (int)floorf((xc + 1.0f) * 2.5f);
  c = c < 0 ? 0 : (c > 4 ? 4 : c);
  float ti = -2.2f + 0.4f * (float)(c + 3);
  float l1 = xc - ti;
  float r1 = (ti + 0.4f) - xc;
  float l2 = l1 + 0.4f, r2 = r1 + 0.4f;
  float l3 = l1 + 0.8f, r3 = r1 + 0.8f;
  float n0 = r1 * 2.5f;
  float n1 = l1 * 2.5f;
  float t0 = n0 * 1.25f;
  float saved = l2 * t0;
  n0 = r1 * t0;
  float t1 = n1 * 1.25f;
  n1 = saved + r2 * t1;
  float n2 = l1 * t1;
  t0 = n0 * (1.0f / 1.2f);
  saved = l3 * t0;
  n0 = r1 * t0;
  t1 = n1 * (1.0f / 1.2f);
  n1 = saved + r2 * t1;
  saved = l2 * t1;
  float t2 = n2 * (1.0f / 1.2f);
  n2 = saved + r3 * t2;
  float n3 = l1 * t2;
  // nonzero window c..c+3 scattered into 4 f16 pairs
  unsigned A = pkh(n0, n1);
  unsigned B = pkh(n2, n3);
  unsigned C = pkh(n1, n2);
  unsigned D = pkh(0.0f, n0);
  unsigned E = pkh(n3, 0.0f);
  p[0] = c == 0 ? A : (c == 1 ? D : 0u);
  p[1] = c == 0 ? B : (c == 1 ? C : (c == 2 ? A : (c == 3 ? D : 0u)));
  p[2] = c == 1 ? E : (c == 2 ? B : (c == 3 ? C : (c == 4 ? A : 0u)));
  p[3] = c == 3 ? E : (c == 4 ? B : 0u);
}

// ---------------------------------------------------------------------------
// TDM descriptor builders + issue (gfx1250 Tensor Data Mover)
// ---------------------------------------------------------------------------
#if __has_builtin(__builtin_amdgcn_tensor_load_to_lds)
#define HAVE_TDM 1
__device__ __forceinline__ void tdm_issue(u32x4 g0, i32x8 g1) {
  i32x4 z4 = {0, 0, 0, 0};
#if __clang_major__ >= 23
  i32x8 z8 = {0, 0, 0, 0, 0, 0, 0, 0};
  __builtin_amdgcn_tensor_load_to_lds(g0, g1, z4, z4, z8, 0);
#else
  __builtin_amdgcn_tensor_load_to_lds(g0, g1, z4, z4, 0);
#endif
}

__device__ __forceinline__ void tdm_group0(u32x4* g0, unsigned lds_off,
                                           const void* gptr) {
  unsigned long long ga = (unsigned long long)(uintptr_t)gptr;
  (*g0)[0] = 1u;                                             // count=1
  (*g0)[1] = lds_off;                                        // lds_addr
  (*g0)[2] = (unsigned)ga;                                   // global lo32
  (*g0)[3] = (unsigned)((ga >> 32) & 0x1FFFFFFu) | (2u << 30); // hi25|type=2
}

// 1D contiguous copy of n dwords (n < 65536): Wpack stage chunk
__device__ __forceinline__ void tdm_load_1d(unsigned lds_off, const void* gptr,
                                            unsigned n) {
  u32x4 g0; tdm_group0(&g0, lds_off, gptr);
  i32x8 g1 = {0, 0, 0, 0, 0, 0, 0, 0};
  g1[0] = (2 << 16);                       // data_size = 4B
  g1[1] = (int)((n & 0xFFFFu) << 16);      // tensor_dim0 lo16
  g1[2] = (int)((n >> 16) & 0xFFFFu) | (1 << 16); // tensor_dim0 hi | dim1=1
  g1[3] = (int)((n & 0xFFFFu) << 16);      // tile_dim0 = n
  g1[4] = 1;                               // tile_dim1 = 1, tile_dim2 = 0
  g1[5] = (int)n;                          // stride0 (unused for 1 row)
  tdm_issue(g0, g1);
}

// 2D x tile: 64 dwords x 128 rows, row stride 512 dwords, LDS pad 4 dw / 64 dw
__device__ __forceinline__ void tdm_load_x(unsigned lds_off, const void* gptr) {
  u32x4 g0; tdm_group0(&g0, lds_off, gptr);
  i32x8 g1 = {0, 0, 0, 0, 0, 0, 0, 0};
  // data_size=4B | pad_enable | pad_interval=5 (64 dw) | pad_amount=3 (4 dw)
  g1[0] = (2 << 16) | (1 << 20) | (5 << 22) | (3 << 25);
  g1[1] = (int)(64u << 16);                // tensor_dim0 = 64
  g1[2] = (int)(128u << 16);               // tensor_dim1 = 128
  g1[3] = (int)(64u << 16);                // tile_dim0 = 64
  g1[4] = 128;                             // tile_dim1 = 128, tile_dim2 = 0
  g1[5] = 512;                             // tensor_dim0_stride = 512 dwords
  tdm_issue(g0, g1);
}
#else
#define HAVE_TDM 0
#endif

// ---------------------------------------------------------------------------
// Main fused kernel: 256 threads (8 waves), block = 128 rows x 64 outputs.
// ---------------------------------------------------------------------------
__global__ void __launch_bounds__(256)
kan_main(const float* __restrict__ x,
         const _Float16* __restrict__ wpack,
         float* __restrict__ out) {
  __shared__ float lds_x[ROWS_PER_BLK * XS];                 // 34.8 KB [row][f]
  __shared__ __align__(32) _Float16 ldsW[CHUNK_HALVES];      // 73.7 KB

  const int tid   = threadIdx.x;
  const int lane  = tid & 31;
  const int wave  = tid >> 5;
  const int half  = lane >> 4;
  const int lrow  = lane & 15;
  const int mbase = wave * 16;
  const int row_l = mbase + lrow;
  const int rbase = blockIdx.x * ROWS_PER_BLK;

  v8f acc[4];
  #pragma unroll
  for (int nt = 0; nt < 4; ++nt)
    acc[nt] = (v8f){0.f, 0.f, 0.f, 0.f, 0.f, 0.f, 0.f, 0.f};

  const v16h* bsrc = (const v16h*)ldsW;
  const unsigned ldsx_off = (unsigned)(uintptr_t)(&lds_x[0]);
  const unsigned ldsw_off = (unsigned)(uintptr_t)(&ldsW[0]);
  (void)ldsx_off; (void)ldsw_off;

  for (int s = 0; s < N_STAGES; ++s) {
    __syncthreads();  // previous-stage consumers done before overwrite
#if HAVE_TDM
    if (wave == 0) {
      tdm_load_x(ldsx_off, x + (size_t)rbase * IN_F + s * 64);
      tdm_load_1d(ldsw_off, wpack + (size_t)s * CHUNK_HALVES, CHUNK_HALVES / 2);
      __builtin_amdgcn_s_wait_tensorcnt(0);
    }
#else
    for (int idx = tid; idx < ROWS_PER_BLK * 64; idx += 256) {
      int r = idx >> 6, f = idx & 63;
      lds_x[r * XS + f] = x[(size_t)(rbase + r) * IN_F + s * 64 + f];
    }
    {
      const uint2* src = (const uint2*)(wpack + (size_t)s * CHUNK_HALVES);
      uint2* dst = (uint2*)ldsW;
      for (int idx = tid; idx < CHUNK_HALVES / 4; idx += 256) dst[idx] = src[idx];
    }
#endif
    __syncthreads();

    // ---- 16 spline k-tiles: 4 features each, 2 basis evals per lane
    for (int t = 0; t < 16; ++t) {
      int fA = t * 4 + half;       // lane<16 -> f0,f2 ; lane>=16 -> f1,f3
      float xA = lds_x[row_l * XS + fA];
      float xB = lds_x[row_l * XS + fA + 2];
      unsigned a0[4], a1[4];
      basis8_pk(xA, a0);
      basis8_pk(xB, a1);
      union { v8u u; v16h h; } af;
      #pragma unroll
      for (int p = 0; p < 4; ++p) { af.u[p] = a0[p]; af.u[p + 4] = a1[p]; }
      #pragma unroll
      for (int nt = 0; nt < 4; ++nt) {
        v16h bf = bsrc[(t * 4 + nt) * 32 + lane];
        acc[nt] = __builtin_amdgcn_wmma_f32_16x16x32_f16(
            false, af.h, false, bf, (short)0, acc[nt], false, false);
      }
    }

    // ---- 2 base k-tiles: 32 silu slots each
    for (int tb = 0; tb < 2; ++tb) {
      int fb = tb * 32;
      union { v8u u; v16h h; } af;
      #pragma unroll
      for (int p = 0; p < 8; ++p) {        // element pair (2p, 2p+1)
        int e0 = 2 * p;
        int K0 = (half ? 8 : 0) + ((e0 < 8) ? e0 : e0 + 8);
        float x0 = lds_x[row_l * XS + fb + K0];
        float x1 = lds_x[row_l * XS + fb + K0 + 1];
        af.u[p] = pkh(silu(x0), silu(x1));
      }
      #pragma unroll
      for (int nt = 0; nt < 4; ++nt) {
        v16h bf = bsrc[((16 + tb) * 4 + nt) * 32 + lane];
        acc[nt] = __builtin_amdgcn_wmma_f32_16x16x32_f16(
            false, af.h, false, bf, (short)0, acc[nt], false, false);
      }
    }
  }

  // ---- store: lane<16: M=r, N=lane ; lane>=16: M=r+8, N=lane-16
  #pragma unroll
  for (int nt = 0; nt < 4; ++nt) {
    #pragma unroll
    for (int r = 0; r < 8; ++r) {
      int rowM = half ? (r + 8) : r;
      out[(size_t)(rbase + mbase + rowM) * OUT_F + nt * 16 + lrow] = acc[nt][r];
    }
  }
}

// ---------------------------------------------------------------------------
extern "C" void kernel_launch(void* const* d_in, const int* in_sizes, int n_in,
                              void* d_out, int out_size, void* d_ws, size_t ws_size,
                              hipStream_t stream) {
  const float* x  = (const float*)d_in[0];   // (32768,512)
  const float* sw = (const float*)d_in[1];   // (512,64,8)
  const float* bw = (const float*)d_in[2];   // (512,64)
  const float* ss = (const float*)d_in[3];   // (512,64)
  float* out = (float*)d_out;                // (32768,64) f32
  _Float16* wpack = (_Float16*)d_ws;         // 576 KB scratch

  const int total = N_KT * 4 * 32 * 16;
  kan_pack_w<<<(total + 255) / 256, 256, 0, stream>>>(sw, bw, ss, wpack);
  kan_main<<<BATCH / ROWS_PER_BLK, 256, 0, stream>>>(x, wpack, out);
}
